// MultiHeadAttention_48550310314073
// MI455X (gfx1250) — compile-verified, matrix operand order checked by reading, not run on hardware
//
#include <hip/hip_runtime.h>
#include <hip/hip_bf16.h>

// ---- problem constants (match reference) ----
constexpr int B  = 2;
constexpr int L  = 2048;
constexpr int DM = 1024;   // D_MODEL
constexpr int H  = 16;
constexpr int DK = 64;
constexpr int DV = 64;
constexpr int NHDK = H * DK;   // 1024

typedef __attribute__((ext_vector_type(16))) _Float16 v16h;
typedef __attribute__((ext_vector_type(8)))  _Float16 v8h;
typedef __attribute__((ext_vector_type(8)))  float    v8f;

// 16-bit WMMA A/B operand layout (16x32): lane holds two contiguous runs of 8
// K-values: [c0, c0+8) and [c0+16, c0+24), where c0 = k0 + 8*(lane>=16).
// Element j of the v16h: j<8 -> k=c0+j ; j>=8 -> k=c0+8+j.

// Load a fragment from a contiguous f16 row: 2x b128 loads.
__device__ __forceinline__ v16h load_frag_f16(const _Float16* __restrict__ rowp, int c0) {
  v8h lo = *(const v8h*)(rowp + c0);
  v8h hi = *(const v8h*)(rowp + c0 + 16);
  v16h r;
#pragma unroll
  for (int j = 0; j < 8; ++j) { r[j] = lo[j]; r[8 + j] = hi[j]; }
  return r;
}

// Load a fragment from a contiguous f32 row: 4x b128 loads + packed cvt.
__device__ __forceinline__ v16h load_frag_f32(const float* __restrict__ rowp, int c0) {
  float4 a0 = *(const float4*)(rowp + c0);
  float4 a1 = *(const float4*)(rowp + c0 + 4);
  float4 b0 = *(const float4*)(rowp + c0 + 16);
  float4 b1 = *(const float4*)(rowp + c0 + 20);
  v16h r;
  r[0]  = (_Float16)a0.x; r[1]  = (_Float16)a0.y; r[2]  = (_Float16)a0.z; r[3]  = (_Float16)a0.w;
  r[4]  = (_Float16)a1.x; r[5]  = (_Float16)a1.y; r[6]  = (_Float16)a1.z; r[7]  = (_Float16)a1.w;
  r[8]  = (_Float16)b0.x; r[9]  = (_Float16)b0.y; r[10] = (_Float16)b0.z; r[11] = (_Float16)b0.w;
  r[12] = (_Float16)b1.x; r[13] = (_Float16)b1.y; r[14] = (_Float16)b1.z; r[15] = (_Float16)b1.w;
  return r;
}

#define WMMA_F16(A, Bm, C) \
  __builtin_amdgcn_wmma_f32_16x16x32_f16(false, (A), false, (Bm), (short)0, (C), false, false)

// ======================================================================
// Kernel 1: projection  Out[b][h][l][d] = sum_c X[b,l,c]*W[h*DK+d,c] + bias
// One wave computes a 16(row) x 64(col) strip = one head's DK for 16 rows.
// ======================================================================
__global__ __launch_bounds__(256)
void proj_kernel(const float* __restrict__ X,     // [B*L, DM]
                 const float* __restrict__ W,     // [NHDK, DM]
                 const float* __restrict__ bias,  // [NHDK]
                 _Float16* __restrict__ Out) {    // [B][H][L][DK]
  const int lane  = threadIdx.x & 31;
  const int gwave = blockIdx.x * (blockDim.x >> 5) + (threadIdx.x >> 5);
  constexpr int CGRP = NHDK / 64;                 // 16 column groups (== heads)
  const int tr = gwave / CGRP;                    // 0..255  (B*L/16)
  const int cg = gwave % CGRP;                    // head index
  const int row0 = tr * 16;
  const int b  = row0 / L;
  const int l0 = row0 % L;
  const int n0g = cg * 64;
  const int mn = lane & 15;
  const int hoff = (lane & 16) ? 8 : 0;

  const float* arow = X + (size_t)(row0 + mn) * DM;
  const float* brow0 = W + (size_t)(n0g      + mn) * DM;
  const float* brow1 = W + (size_t)(n0g + 16 + mn) * DM;
  const float* brow2 = W + (size_t)(n0g + 32 + mn) * DM;
  const float* brow3 = W + (size_t)(n0g + 48 + mn) * DM;

  v8f acc0 = {}, acc1 = {}, acc2 = {}, acc3 = {};
  for (int k0 = 0; k0 < DM; k0 += 32) {
    const int c0 = k0 + hoff;
    v16h a  = load_frag_f32(arow,  c0);
    v16h b0 = load_frag_f32(brow0, c0);
    v16h b1 = load_frag_f32(brow1, c0);
    v16h b2 = load_frag_f32(brow2, c0);
    v16h b3 = load_frag_f32(brow3, c0);
    acc0 = WMMA_F16(a, b0, acc0);
    acc1 = WMMA_F16(a, b1, acc1);
    acc2 = WMMA_F16(a, b2, acc2);
    acc3 = WMMA_F16(a, b3, acc3);
  }

  const int mbase = (lane & 16) ? 8 : 0;
  const size_t rowb = (size_t)(b * H + cg) * L;   // h == cg
  v8f* accs[4] = { &acc0, &acc1, &acc2, &acc3 };
#pragma unroll
  for (int t = 0; t < 4; ++t) {
    const float bv = bias[n0g + 16 * t + mn];
    const int d0 = 16 * t;
#pragma unroll
    for (int r = 0; r < 8; ++r) {
      Out[(rowb + (l0 + r + mbase)) * DK + d0 + mn] = (_Float16)((*accs[t])[r] + bv);
    }
  }
}

// ======================================================================
// Kernel 1b: transpose V -> Vt[b][dv][k] in f16 (so attn@V B-fragments
// become contiguous b128 loads).
// ======================================================================
__global__ __launch_bounds__(256)
void vt_kernel(const float* __restrict__ V,       // [B][L][DV]
               _Float16* __restrict__ Vt) {       // [B][DV][L]
  const int idx = blockIdx.x * 256 + threadIdx.x; // over B*DV*L
  const int k  = idx % L;
  const int bd = idx / L;
  const int d  = bd % DV;
  const int b  = bd / DV;
  Vt[idx] = (_Float16)V[((size_t)b * L + k) * DV + d];
}

// ======================================================================
// Kernel 2: raw scaled scores  S[h*B+b, q, k] = scale * Qh(q,:) . Kh(k,:)
// One wave computes a 32(q) x 64(k) strip: 2 Q frags x 4 K frags,
// 8 accumulators, 8 WMMAs per K-step (hides the WMMA->VALU WAR NOPs).
// ======================================================================
__global__ __launch_bounds__(256)
void score_kernel(const _Float16* __restrict__ Qh,  // [B][H][L][DK]
                  const _Float16* __restrict__ Kh,  // [B][H][L][DK]
                  float* __restrict__ attn) {       // [H*B][L][L]
  const int lane  = threadIdx.x & 31;
  const int gwave = blockIdx.x * (blockDim.x >> 5) + (threadIdx.x >> 5);
  constexpr int TQG = L / 32;                      // 64 q-groups
  constexpr int TKG = L / 64;                      // 32 k-groups
  constexpr int TILES_HB = TQG * TKG;              // 2048
  const int hb  = gwave / TILES_HB;                // h*B + b
  const int rem = gwave % TILES_HB;
  const int tqg = rem / TKG, tkg = rem % TKG;
  const int h = hb / B, b = hb % B;
  const int q0 = tqg * 32, k0g = tkg * 64;
  const int mn = lane & 15;
  const int hoff = (lane & 16) ? 8 : 0;

  const size_t head = (size_t)(b * H + h) * L;
  const _Float16* arowA = Qh + (head + q0      + mn) * DK;
  const _Float16* arowB = Qh + (head + q0 + 16 + mn) * DK;
  const _Float16* brow0 = Kh + (head + k0g      + mn) * DK;
  const _Float16* brow1 = Kh + (head + k0g + 16 + mn) * DK;
  const _Float16* brow2 = Kh + (head + k0g + 32 + mn) * DK;
  const _Float16* brow3 = Kh + (head + k0g + 48 + mn) * DK;

  v8f acc[2][4] = {};
#pragma unroll
  for (int kk = 0; kk < DK; kk += 32) {
    const int c0 = kk + hoff;
    v16h aA = load_frag_f16(arowA, c0);
    v16h aB = load_frag_f16(arowB, c0);
    v16h b0 = load_frag_f16(brow0, c0);
    v16h b1 = load_frag_f16(brow1, c0);
    v16h b2 = load_frag_f16(brow2, c0);
    v16h b3 = load_frag_f16(brow3, c0);
    acc[0][0] = WMMA_F16(aA, b0, acc[0][0]);
    acc[1][0] = WMMA_F16(aB, b0, acc[1][0]);
    acc[0][1] = WMMA_F16(aA, b1, acc[0][1]);
    acc[1][1] = WMMA_F16(aB, b1, acc[1][1]);
    acc[0][2] = WMMA_F16(aA, b2, acc[0][2]);
    acc[1][2] = WMMA_F16(aB, b2, acc[1][2]);
    acc[0][3] = WMMA_F16(aA, b3, acc[0][3]);
    acc[1][3] = WMMA_F16(aB, b3, acc[1][3]);
  }

  const float scale = 0.125f;                      // 1/sqrt(64)
  const int mbase = (lane & 16) ? 8 : 0;
#pragma unroll
  for (int s = 0; s < 2; ++s) {
#pragma unroll
    for (int t = 0; t < 4; ++t) {
#pragma unroll
      for (int r = 0; r < 8; ++r) {
        const size_t idx = ((size_t)hb * L + (q0 + 16 * s + r + mbase)) * L + k0g + 16 * t + mn;
        attn[idx] = acc[s][t][r] * scale;
      }
    }
  }
}

// ======================================================================
// Kernel 3: row softmax in place. One 256-thread block per row of 2048.
// ======================================================================
__global__ __launch_bounds__(256)
void softmax_kernel(float* __restrict__ attn) {
  const int tid = threadIdx.x;
  float* row = attn + (size_t)blockIdx.x * L;

  float vals[8];
  float mx = -3.402823466e38f;
#pragma unroll
  for (int i = 0; i < 8; ++i) {
    vals[i] = row[tid + i * 256];
    mx = fmaxf(mx, vals[i]);
  }
#pragma unroll
  for (int off = 16; off > 0; off >>= 1)
    mx = fmaxf(mx, __shfl_xor(mx, off, 32));

  __shared__ float red_max[8];
  __shared__ float red_sum[8];
  const int wid = tid >> 5;
  if ((tid & 31) == 0) red_max[wid] = mx;
  __syncthreads();
  float m_all = red_max[0];
#pragma unroll
  for (int i = 1; i < 8; ++i) m_all = fmaxf(m_all, red_max[i]);

  float s = 0.0f;
#pragma unroll
  for (int i = 0; i < 8; ++i) {
    vals[i] = __expf(vals[i] - m_all);
    s += vals[i];
  }
#pragma unroll
  for (int off = 16; off > 0; off >>= 1)
    s += __shfl_xor(s, off, 32);
  if ((tid & 31) == 0) red_sum[wid] = s;
  __syncthreads();
  float total = 0.0f;
#pragma unroll
  for (int i = 0; i < 8; ++i) total += red_sum[i];

  const float inv = 1.0f / total;
#pragma unroll
  for (int i = 0; i < 8; ++i) row[tid + i * 256] = vals[i] * inv;
}

// ======================================================================
// Kernel 4: out[b,q,h*DV+dv] = sum_k attn[h*B+b,q,k] * v[b,k,dv]
// One wave computes a 64(q) x 32(dv) strip: 4 attn frags x 2 Vt frags,
// 8 accumulators; all operands are contiguous b128 loads (Vt transposed).
// ======================================================================
__global__ __launch_bounds__(256)
void out_kernel(const float* __restrict__ attn,   // [H*B][L][L]
                const _Float16* __restrict__ Vt,  // [B][DV][L]
                float* __restrict__ out) {        // [B][L][H*DV]
  const int lane  = threadIdx.x & 31;
  const int gwave = blockIdx.x * (blockDim.x >> 5) + (threadIdx.x >> 5);
  constexpr int TQG = L / 64;                     // 32 q-groups
  constexpr int TCG = (H * DV) / 32;              // 32 col-groups
  constexpr int TILES_B = TQG * TCG;              // 1024
  const int b   = gwave / TILES_B;
  const int rem = gwave % TILES_B;
  const int tqg = rem / TCG, tcg = rem % TCG;
  const int d0g = tcg * 32;                       // global out-col base
  const int h   = d0g / DV;                       // head for this col group
  const int dv0 = d0g % DV;                       // dv base within head
  const int q0g = tqg * 64;
  const int mn = lane & 15;
  const int hoff = (lane & 16) ? 8 : 0;

  const size_t hbrow = (size_t)(h * B + b) * L;
  const float* arow0 = attn + (hbrow + q0g      + mn) * L;
  const float* arow1 = attn + (hbrow + q0g + 16 + mn) * L;
  const float* arow2 = attn + (hbrow + q0g + 32 + mn) * L;
  const float* arow3 = attn + (hbrow + q0g + 48 + mn) * L;
  const _Float16* vrow0 = Vt + ((size_t)b * DV + dv0      + mn) * L;
  const _Float16* vrow1 = Vt + ((size_t)b * DV + dv0 + 16 + mn) * L;

  v8f acc[4][2] = {};
  for (int k0 = 0; k0 < L; k0 += 32) {
    const int c0 = k0 + hoff;
    v16h b0 = load_frag_f16(vrow0, c0);
    v16h b1 = load_frag_f16(vrow1, c0);
    v16h a0 = load_frag_f32(arow0, c0);
    v16h a1 = load_frag_f32(arow1, c0);
    v16h a2 = load_frag_f32(arow2, c0);
    v16h a3 = load_frag_f32(arow3, c0);
    acc[0][0] = WMMA_F16(a0, b0, acc[0][0]);
    acc[0][1] = WMMA_F16(a0, b1, acc[0][1]);
    acc[1][0] = WMMA_F16(a1, b0, acc[1][0]);
    acc[1][1] = WMMA_F16(a1, b1, acc[1][1]);
    acc[2][0] = WMMA_F16(a2, b0, acc[2][0]);
    acc[2][1] = WMMA_F16(a2, b1, acc[2][1]);
    acc[3][0] = WMMA_F16(a3, b0, acc[3][0]);
    acc[3][1] = WMMA_F16(a3, b1, acc[3][1]);
  }

  const int mbase = (lane & 16) ? 8 : 0;
#pragma unroll
  for (int t = 0; t < 4; ++t) {
#pragma unroll
    for (int u = 0; u < 2; ++u) {
#pragma unroll
      for (int r = 0; r < 8; ++r) {
        out[((size_t)b * L + q0g + 16 * t + r + mbase) * (H * DV) + d0g + 16 * u + mn] = acc[t][u][r];
      }
    }
  }
}

// ======================================================================
extern "C" void kernel_launch(void* const* d_in, const int* in_sizes, int n_in,
                              void* d_out, int out_size, void* d_ws, size_t ws_size,
                              hipStream_t stream) {
  const float* q  = (const float*)d_in[0];
  const float* k  = (const float*)d_in[1];
  const float* v  = (const float*)d_in[2];
  const float* Wq = (const float*)d_in[3];
  const float* bq = (const float*)d_in[4];
  const float* Wk = (const float*)d_in[5];
  const float* bk = (const float*)d_in[6];
  (void)in_sizes; (void)n_in; (void)out_size; (void)ws_size;

  float* out  = (float*)d_out;
  float* attn = out + (size_t)B * L * H * DV;       // second output region

  _Float16* Qh = (_Float16*)d_ws;                   // 8 MB
  _Float16* Kh = Qh + (size_t)B * H * L * DK;       // 8 MB
  _Float16* Vt = Kh + (size_t)B * H * L * DK;       // 512 KB

  const dim3 blk(256);                              // 8 wave32 per block

  // projections: 256 row-tiles * 16 col-groups = 4096 waves / 8
  proj_kernel<<<512, blk, 0, stream>>>(q, Wq, bq, Qh);
  proj_kernel<<<512, blk, 0, stream>>>(k, Wk, bk, Kh);

  // V transpose to f16 [b][dv][k]: B*DV*L = 262144 elements
  vt_kernel<<<1024, blk, 0, stream>>>(v, Vt);

  // scores: 32 * 64 * 32 = 65536 waves / 8
  score_kernel<<<8192, blk, 0, stream>>>(Qh, Kh, attn);

  // softmax: one block per row, H*B*L = 65536 rows
  softmax_kernel<<<65536, blk, 0, stream>>>(attn);

  // output GEMM: 2 * 32 * 32 = 2048 waves / 8
  out_kernel<<<256, blk, 0, stream>>>(attn, Vt, out);
}
